// Attention_7146825580674
// MI455X (gfx1250) — compile-verified, matrix operand order checked by reading, not run on hardware
//
#include <hip/hip_runtime.h>
#include <hip/hip_bf16.h>

// ---------------------------------------------------------------------------
// Causal attention, B=4, T=2048, C=1024 (fp32 in/out) via bf16 WMMA on gfx1250.
//  - All GEMMs: block tile 256x64, 8 waves, wave tile 64x32 (4x2 WMMA 16x16x32)
//  - Global->LDS staging uses GLOBAL_LOAD_ASYNC_TO_LDS_B128 (ASYNCcnt) with
//    double-buffered LDS: copy of tile k+1 overlaps WMMA on tile k.
//  - Intermediates (~166 MB) fit MI455X's 192 MB L2 -> two-pass softmax design
//    costs ~no HBM traffic; the problem is compute-bound (~120 GFLOP).
// ---------------------------------------------------------------------------

typedef __attribute__((ext_vector_type(16))) __bf16 v16bf;
typedef __attribute__((ext_vector_type(8)))  float  v8f;
typedef __attribute__((ext_vector_type(4)))  float  f4;
typedef __attribute__((ext_vector_type(4)))  unsigned short us4;

#define B_DIM 4
#define T_DIM 2048
#define C_DIM 1024

// fp32 -> bf16 (round to nearest even), bit-level
__device__ __forceinline__ unsigned short f32_to_bf16(float f) {
  unsigned int u = __builtin_bit_cast(unsigned int, f);
  unsigned int r = u + 0x7FFFu + ((u >> 16) & 1u);
  return (unsigned short)(r >> 16);
}

// Low 32 bits of a generic pointer to __shared__ = LDS byte address.
__device__ __forceinline__ unsigned lds_addr(const void* p) {
  return (unsigned)(unsigned long long)p;
}

// Async 16B global->LDS copy (per-lane), tracked with ASYNCcnt.
// GVS addressing: mem = SADDR(64b SGPR pair) + VADDR(32b byte offset).
__device__ __forceinline__ void async_b128(unsigned lds_byte,
                                           const unsigned short* sbase,
                                           unsigned gbyte) {
  asm volatile("global_load_async_to_lds_b128 %0, %1, %2"
               :
               : "v"(lds_byte), "v"(gbyte), "s"(sbase)
               : "memory");
}

__device__ __forceinline__ void wait_async0() {
  asm volatile("s_wait_asynccnt 0x0" ::: "memory");
}

// Fragment with contraction in the SECOND (contiguous) LDS dim: elem = M[rc][k]
// ISA 16-bit A layout: lanes 0-15: e0..7->K0..7, e8..15->K16..23;
//                      lanes 16-31: e0..7->K8..15, e8..15->K24..31.
__device__ __forceinline__ v16bf frag_k2(const unsigned short* base, int ld,
                                         int rc_base, int lane) {
  const int rc = rc_base + (lane & 15);
  const int kg = (lane >> 4) * 8;
  v16bf f;
#pragma unroll
  for (int e = 0; e < 16; ++e) {
    int k = kg + (e < 8 ? e : e + 8);
    f[e] = __builtin_bit_cast(__bf16, base[rc * ld + k]);
  }
  return f;
}

// Fragment with contraction in the FIRST LDS dim: elem = M[k][n]
__device__ __forceinline__ v16bf frag_k1(const unsigned short* base, int ld,
                                         int n_base, int lane) {
  const int n = n_base + (lane & 15);
  const int kg = (lane >> 4) * 8;
  v16bf f;
#pragma unroll
  for (int e = 0; e < 16; ++e) {
    int k = kg + (e < 8 ? e : e + 8);
    f[e] = __builtin_bit_cast(__bf16, base[k * ld + n]);
  }
  return f;
}

__device__ __forceinline__ v8f wmma_bf16(v16bf a, v16bf b, v8f c) {
  return __builtin_amdgcn_wmma_f32_16x16x32_bf16(
      false, a, false, b, (short)0, c, false, false);
}

// ---------------------------------------------------------------------------
// Kernel 0: fp32 -> bf16 bulk convert (x and the three weight matrices)
// ---------------------------------------------------------------------------
__global__ __launch_bounds__(256) void cvt_bf16_kernel(
    const float* __restrict__ in, unsigned short* __restrict__ out) {
  const int i = (blockIdx.x * 256 + threadIdx.x) * 4;  // sizes divisible by 1024
  f4 v = *(const f4*)(in + i);
  us4 o;
  o.x = f32_to_bf16(v.x);
  o.y = f32_to_bf16(v.y);
  o.z = f32_to_bf16(v.z);
  o.w = f32_to_bf16(v.w);
  *(us4*)(out + i) = o;
}

// ---------------------------------------------------------------------------
// Kernel 1: K/Q/V = x @ W^T + b (contraction on second index of both operands)
// grid.z selects projection. Async double-buffered staging.
// ---------------------------------------------------------------------------
__global__ __launch_bounds__(256) void qkv_proj_kernel(
    const unsigned short* __restrict__ xb,
    const unsigned short* __restrict__ Wkb,
    const unsigned short* __restrict__ Wqb,
    const unsigned short* __restrict__ Wvb,
    const float* __restrict__ bk, const float* __restrict__ bq,
    const float* __restrict__ bv,
    unsigned short* __restrict__ Kb, unsigned short* __restrict__ Qb,
    unsigned short* __restrict__ Vb) {
  const int proj = blockIdx.z;
  const unsigned short* W = (proj == 0) ? Wkb : (proj == 1) ? Wqb : Wvb;
  const float* bias       = (proj == 0) ? bk  : (proj == 1) ? bq  : bv;
  unsigned short* out     = (proj == 0) ? Kb  : (proj == 1) ? Qb  : Vb;

  const int mbase = blockIdx.x * 256;
  const int nbase = blockIdx.y * 64;

  __shared__ __align__(16) unsigned short As[2][256][40];  // 80B rows (16B mult)
  __shared__ __align__(16) unsigned short Bs[2][64][40];

  const int tid  = threadIdx.x;
  const int lane = tid & 31;
  const int wave = tid >> 5;
  const int wm   = (wave >> 1) * 64;   // 0,64,128,192
  const int wn   = (wave & 1) * 32;    // 0,32

  auto stage = [&](int buf, int kb) {
#pragma unroll
    for (int s = 0; s < 4; ++s) {       // 256x32 tile = 1024 x 16B chunks
      int slot = tid + s * 256;
      int r = slot >> 2, ch = slot & 3;
      async_b128(lds_addr(&As[buf][r][ch * 8]), xb,
                 (unsigned)(((mbase + r) * C_DIM + kb + ch * 8) * 2));
    }
    {                                   // 64x32 tile = 256 chunks
      int r = tid >> 2, ch = tid & 3;
      async_b128(lds_addr(&Bs[buf][r][ch * 8]), W,
                 (unsigned)(((nbase + r) * C_DIM + kb + ch * 8) * 2));
    }
  };

  v8f acc[4][2] = {};
  stage(0, 0);
  const int NK = C_DIM / 32;
  for (int t = 0; t < NK; ++t) {
    wait_async0();
    __syncthreads();
    if (t + 1 < NK) stage((t + 1) & 1, (t + 1) * 32);
    const int cur = t & 1;
    v16bf b0 = frag_k2(&Bs[cur][0][0], 40, wn, lane);
    v16bf b1 = frag_k2(&Bs[cur][0][0], 40, wn + 16, lane);
#pragma unroll
    for (int i = 0; i < 4; ++i) {
      v16bf a = frag_k2(&As[cur][0][0], 40, wm + i * 16, lane);
      acc[i][0] = wmma_bf16(a, b0, acc[i][0]);
      acc[i][1] = wmma_bf16(a, b1, acc[i][1]);
    }
  }

  const int coll  = lane & 15;
  const int rowhi = (lane >> 4) * 8;
#pragma unroll
  for (int i = 0; i < 4; ++i)
#pragma unroll
    for (int j = 0; j < 2; ++j) {
      const int n   = nbase + wn + j * 16 + coll;
      const float bb = bias[n];
#pragma unroll
      for (int r = 0; r < 8; ++r) {
        const int m = mbase + wm + i * 16 + rowhi + r;
        out[(size_t)m * C_DIM + n] = f32_to_bf16(acc[i][j][r] + bb);
      }
    }
}

// ---------------------------------------------------------------------------
// Kernel 2: S[b,t,s] = dot(K[b,t,:], Q[b,s,:]) / 32, causal mask (s<=t).
// ---------------------------------------------------------------------------
__global__ __launch_bounds__(256) void scores_kernel(
    const unsigned short* __restrict__ Kb, const unsigned short* __restrict__ Qb,
    float* __restrict__ S) {
  const int b     = blockIdx.z;
  const int tbase = blockIdx.x * 256;
  const int sbase = blockIdx.y * 64;
  float* Sb = S + (size_t)b * T_DIM * T_DIM;
  const int tid = threadIdx.x;

  if (sbase > tbase + 255) {            // fully-masked tile
    for (int i = tid; i < 256 * 64; i += 256) {
      int r = i >> 6, c = i & 63;
      Sb[(size_t)(tbase + r) * T_DIM + sbase + c] = -__builtin_inff();
    }
    return;
  }

  const unsigned short* Kp = Kb + (size_t)b * T_DIM * C_DIM;
  const unsigned short* Qp = Qb + (size_t)b * T_DIM * C_DIM;

  __shared__ __align__(16) unsigned short As[2][256][40];
  __shared__ __align__(16) unsigned short Bs[2][64][40];

  const int lane = tid & 31;
  const int wave = tid >> 5;
  const int wm   = (wave >> 1) * 64;
  const int wn   = (wave & 1) * 32;

  auto stage = [&](int buf, int kb) {
#pragma unroll
    for (int s = 0; s < 4; ++s) {
      int slot = tid + s * 256;
      int r = slot >> 2, ch = slot & 3;
      async_b128(lds_addr(&As[buf][r][ch * 8]), Kp,
                 (unsigned)(((tbase + r) * C_DIM + kb + ch * 8) * 2));
    }
    {
      int r = tid >> 2, ch = tid & 3;
      async_b128(lds_addr(&Bs[buf][r][ch * 8]), Qp,
                 (unsigned)(((sbase + r) * C_DIM + kb + ch * 8) * 2));
    }
  };

  v8f acc[4][2] = {};
  stage(0, 0);
  const int NK = C_DIM / 32;
  for (int t = 0; t < NK; ++t) {
    wait_async0();
    __syncthreads();
    if (t + 1 < NK) stage((t + 1) & 1, (t + 1) * 32);
    const int cur = t & 1;
    v16bf b0 = frag_k2(&Bs[cur][0][0], 40, wn, lane);
    v16bf b1 = frag_k2(&Bs[cur][0][0], 40, wn + 16, lane);
#pragma unroll
    for (int i = 0; i < 4; ++i) {
      v16bf a = frag_k2(&As[cur][0][0], 40, wm + i * 16, lane);
      acc[i][0] = wmma_bf16(a, b0, acc[i][0]);
      acc[i][1] = wmma_bf16(a, b1, acc[i][1]);
    }
  }

  const float scale = 0.03125f;          // 1/sqrt(1024)
  const int coll  = lane & 15;
  const int rowhi = (lane >> 4) * 8;
#pragma unroll
  for (int i = 0; i < 4; ++i)
#pragma unroll
    for (int j = 0; j < 2; ++j)
#pragma unroll
      for (int r = 0; r < 8; ++r) {
        const int t2 = tbase + wm + i * 16 + rowhi + r;
        const int s2 = sbase + wn + j * 16 + coll;
        const float v = acc[i][j][r] * scale;
        Sb[(size_t)t2 * T_DIM + s2] = (s2 <= t2) ? v : -__builtin_inff();
      }
}

// ---------------------------------------------------------------------------
// Kernel 3: row softmax (one 256-thread block per row of 2048), writes bf16 P
// ---------------------------------------------------------------------------
__global__ __launch_bounds__(256) void softmax_kernel(
    const float* __restrict__ S, unsigned short* __restrict__ P) {
  const int row = blockIdx.x;                       // b*T + t
  const float* s = S + (size_t)row * T_DIM;
  unsigned short* p = P + (size_t)row * T_DIM;
  const int tid = threadIdx.x;

  __shared__ float red[256];
  float v[8];
  float m = -__builtin_inff();
#pragma unroll
  for (int i = 0; i < 8; ++i) {
    v[i] = s[tid + i * 256];
    m = fmaxf(m, v[i]);
  }
  red[tid] = m;
  __syncthreads();
  for (int off = 128; off > 0; off >>= 1) {
    if (tid < off) red[tid] = fmaxf(red[tid], red[tid + off]);
    __syncthreads();
  }
  m = red[0];
  __syncthreads();

  float sum = 0.0f;
#pragma unroll
  for (int i = 0; i < 8; ++i) {
    v[i] = __expf(v[i] - m);                        // exp(-inf)=0 for masked
    sum += v[i];
  }
  red[tid] = sum;
  __syncthreads();
  for (int off = 128; off > 0; off >>= 1) {
    if (tid < off) red[tid] += red[tid + off];
    __syncthreads();
  }
  const float inv = 1.0f / red[0];
#pragma unroll
  for (int i = 0; i < 8; ++i) p[tid + i * 256] = f32_to_bf16(v[i] * inv);
}

// ---------------------------------------------------------------------------
// Kernel 4: O[b,t,c] = sum_s P[b,t,s] * V[b,s,c]; k-loop bounded by causality.
// ---------------------------------------------------------------------------
__global__ __launch_bounds__(256) void pv_kernel(
    const unsigned short* __restrict__ P, const unsigned short* __restrict__ Vb,
    float* __restrict__ O) {
  const int b     = blockIdx.z;
  const int tbase = blockIdx.x * 256;
  const int nbase = blockIdx.y * 64;
  const unsigned short* Pp = P + (size_t)b * T_DIM * T_DIM;
  const unsigned short* Vp = Vb + (size_t)b * T_DIM * C_DIM;
  float* Op = O + (size_t)b * T_DIM * C_DIM;

  __shared__ __align__(16) unsigned short Ps[2][256][40];
  __shared__ __align__(16) unsigned short Vs[2][32][72];  // 144B rows

  const int tid  = threadIdx.x;
  const int lane = tid & 31;
  const int wave = tid >> 5;
  const int wm   = (wave >> 1) * 64;
  const int wn   = (wave & 1) * 32;

  auto stage = [&](int buf, int kb) {
#pragma unroll
    for (int s = 0; s < 4; ++s) {
      int slot = tid + s * 256;
      int r = slot >> 2, ch = slot & 3;
      async_b128(lds_addr(&Ps[buf][r][ch * 8]), Pp,
                 (unsigned)(((tbase + r) * T_DIM + kb + ch * 8) * 2));
    }
    {                                    // 32x64 tile = 256 chunks (8 per row)
      int r = tid >> 3, ch = tid & 7;
      async_b128(lds_addr(&Vs[buf][r][ch * 8]), Vp,
                 (unsigned)(((kb + r) * C_DIM + nbase + ch * 8) * 2));
    }
  };

  v8f acc[4][2] = {};
  stage(0, 0);
  const int NK = (tbase + 256) / 32;     // P[t,s]==0 for s>t
  for (int t = 0; t < NK; ++t) {
    wait_async0();
    __syncthreads();
    if (t + 1 < NK) stage((t + 1) & 1, (t + 1) * 32);
    const int cur = t & 1;
    v16bf b0 = frag_k1(&Vs[cur][0][0], 72, wn, lane);
    v16bf b1 = frag_k1(&Vs[cur][0][0], 72, wn + 16, lane);
#pragma unroll
    for (int i = 0; i < 4; ++i) {
      v16bf a = frag_k2(&Ps[cur][0][0], 40, wm + i * 16, lane);
      acc[i][0] = wmma_bf16(a, b0, acc[i][0]);
      acc[i][1] = wmma_bf16(a, b1, acc[i][1]);
    }
  }

  const int coll  = lane & 15;
  const int rowhi = (lane >> 4) * 8;
#pragma unroll
  for (int i = 0; i < 4; ++i)
#pragma unroll
    for (int j = 0; j < 2; ++j)
#pragma unroll
      for (int r = 0; r < 8; ++r) {
        const int t2 = tbase + wm + i * 16 + rowhi + r;
        const int n  = nbase + wn + j * 16 + coll;
        Op[(size_t)t2 * C_DIM + n] = acc[i][j][r];
      }
}

// ---------------------------------------------------------------------------
extern "C" void kernel_launch(void* const* d_in, const int* in_sizes, int n_in,
                              void* d_out, int out_size, void* d_ws,
                              size_t ws_size, hipStream_t stream) {
  const float* x  = (const float*)d_in[0];
  const float* Wk = (const float*)d_in[1];
  const float* Wq = (const float*)d_in[2];
  const float* Wv = (const float*)d_in[3];
  const float* bk = (const float*)d_in[4];
  const float* bq = (const float*)d_in[5];
  const float* bv = (const float*)d_in[6];
  float* out = (float*)d_out;

  // Workspace layout (166 MB total, fits MI455X 192MB L2):
  //  xb 16MB | Wkb/Wqb/Wvb 2MB ea | Kb/Qb/Vb 16MB ea | S 64MB | P 32MB
  const size_t x_elems   = (size_t)B_DIM * T_DIM * C_DIM;  // 8M
  const size_t w_elems   = (size_t)C_DIM * C_DIM;          // 1M
  const size_t s_elems   = (size_t)B_DIM * T_DIM * T_DIM;  // 16M
  unsigned short* xb  = (unsigned short*)d_ws;
  unsigned short* Wkb = xb + x_elems;
  unsigned short* Wqb = Wkb + w_elems;
  unsigned short* Wvb = Wqb + w_elems;
  unsigned short* Kb  = Wvb + w_elems;
  unsigned short* Qb  = Kb + x_elems;
  unsigned short* Vb  = Qb + x_elems;
  float*          S   = (float*)(Vb + x_elems);
  unsigned short* P   = (unsigned short*)(S + s_elems);

  const int M = B_DIM * T_DIM;                             // 8192

  cvt_bf16_kernel<<<dim3((int)(x_elems / 1024)), 256, 0, stream>>>(x, xb);
  cvt_bf16_kernel<<<dim3((int)(w_elems / 1024)), 256, 0, stream>>>(Wk, Wkb);
  cvt_bf16_kernel<<<dim3((int)(w_elems / 1024)), 256, 0, stream>>>(Wq, Wqb);
  cvt_bf16_kernel<<<dim3((int)(w_elems / 1024)), 256, 0, stream>>>(Wv, Wvb);

  qkv_proj_kernel<<<dim3(M / 256, C_DIM / 64, 3), 256, 0, stream>>>(
      xb, Wkb, Wqb, Wvb, bk, bq, bv, Kb, Qb, Vb);
  scores_kernel<<<dim3(T_DIM / 256, T_DIM / 64, B_DIM), 256, 0, stream>>>(
      Kb, Qb, S);
  softmax_kernel<<<dim3(M, 1, 1), 256, 0, stream>>>(S, P);
  pv_kernel<<<dim3(T_DIM / 256, C_DIM / 64, B_DIM), 256, 0, stream>>>(
      P, Vb, out);
}
